// MixAttention_58944131170462
// MI455X (gfx1250) — matrix operand order repaired and checked by reading, not compiled, not run on hardware
//
#include <hip/hip_runtime.h>
#include <hip/hip_bf16.h>
#include <math.h>

#define N_NODES 8192
#define IN_FEAT 256
#define OUT_FEAT 128
#define N_EDGES 262144
#define ADJ_WORDS 256          // 8192 bits / 32 per row
#define SEG_WORDS 64
#define CAP (SEG_WORDS * 32)   // 2048 entries, 16KB LDS

typedef __attribute__((ext_vector_type(16))) _Float16 v16h;
typedef __attribute__((ext_vector_type(8)))  float    v8f;

// ---------------- wave32 reductions ----------------
__device__ inline float waveSum(float v) {
#pragma unroll
  for (int m = 16; m > 0; m >>= 1) v += __shfl_xor(v, m, 32);
  return v;
}
__device__ inline float waveMax(float v) {
#pragma unroll
  for (int m = 16; m > 0; m >>= 1) v = fmaxf(v, __shfl_xor(v, m, 32));
  return v;
}

// ---------------- fold Ws_w / as_w into u1,u2,bu ----------------
__global__ void k_u_vectors(const float* __restrict__ Ws_w, const float* __restrict__ Ws_b,
                            const float* __restrict__ as_w,
                            float* __restrict__ u1, float* __restrict__ u2,
                            float* __restrict__ bu) {
  int k = threadIdx.x;
  if (k < IN_FEAT) {
    float s1 = 0.f, s2 = 0.f;
    for (int o = 0; o < OUT_FEAT; ++o) {
      float w = Ws_w[(size_t)o * IN_FEAT + k];
      s1 += w * as_w[o];
      s2 += w * as_w[OUT_FEAT + o];
    }
    u1[k] = s1; u2[k] = s2;
  }
  if (k == 0) {
    float b1 = 0.f, b2 = 0.f;
    for (int o = 0; o < OUT_FEAT; ++o) {
      b1 += Ws_b[o] * as_w[o];
      b2 += Ws_b[o] * as_w[OUT_FEAT + o];
    }
    bu[0] = b1; bu[1] = b2;
  }
}

// ---------------- hc = h_context @ Wc_w^T + b via WMMA ----------------
// grid = (512, 8), block = 32 (one wave per 16x16 output tile)
__global__ void k_gemm_hc(const float* __restrict__ X, const float* __restrict__ W,
                          const float* __restrict__ bias, float* __restrict__ hc) {
  const int lane = threadIdx.x;      // 0..31
  const int half = lane >> 4;        // 0 or 1
  const int l16  = lane & 15;
  const int tm = blockIdx.x * 16;    // node tile
  const int tn = blockIdx.y * 16;    // out-feature tile

  v8f acc = {};
#pragma unroll
  for (int k0 = 0; k0 < IN_FEAT; k0 += 32) {
    v16h a, b;
    // A (16x32 f16): lane holds row M = tm+l16.
    // element e -> VGPR v=e/2: v<4: K = 8*half + e ; v>=4: K = 16 + 8*half + (e-8)
    const float* arow = X + (size_t)(tm + l16) * IN_FEAT + k0;
#pragma unroll
    for (int e = 0; e < 16; ++e) {
      int k = (e < 8) ? (half * 8 + e) : (16 + half * 8 + (e - 8));
      a[e] = (_Float16)arow[k];
    }
    // B (32x16 f16): column n = tn+l16; same K order as A: e<8: K = 8*half + e ; e>=8: K = 16 + 8*half + (e-8).
    // B[k][n] = W[n*IN_FEAT + k]  (Wc_w is [OUT_FEAT, IN_FEAT] row-major)
    const float* wrow = W + (size_t)(tn + l16) * IN_FEAT + k0;
#pragma unroll
    for (int e = 0; e < 16; ++e) b[e] = (_Float16)wrow[(e < 8) ? (half * 8 + e) : (16 + half * 8 + (e - 8))];

    acc = __builtin_amdgcn_wmma_f32_16x16x32_f16(
        /*neg_a=*/false, a, /*neg_b=*/false, b,
        /*c_mod=*/(short)0, acc, /*reuse_a=*/false, /*reuse_b=*/false);
  }
  // D layout: VGPR r -> M = tm + 8*half + r, N = tn + l16
  float bn = bias[tn + l16];
#pragma unroll
  for (int r = 0; r < 8; ++r) {
    int m = tm + half * 8 + r;
    hc[(size_t)m * OUT_FEAT + tn + l16] = acc[r] + bn;
  }
}

// ---------------- context scores: c_i = hc_i . a1 ; d_i = hc_i . a2 ----------------
__global__ void k_ctx_scores(const float* __restrict__ hc, const float* __restrict__ ac_w,
                             float* __restrict__ cvec, float* __restrict__ dvec) {
  const int wave = threadIdx.x >> 5, lane = threadIdx.x & 31;
  const int row = blockIdx.x * 8 + wave;
  const float* h = hc + (size_t)row * OUT_FEAT;
  float s1 = 0.f, s2 = 0.f;
#pragma unroll
  for (int o = lane; o < OUT_FEAT; o += 32) {
    float v = h[o];
    s1 += v * ac_w[o];
    s2 += v * ac_w[OUT_FEAT + o];
  }
  s1 = waveSum(s1); s2 = waveSum(s2);
  if (lane == 0) { cvec[row] = s1; dvec[row] = s2; }
}

// ---------------- structure scores: e_i = softmax(hstr_i).u1 + bu1 ----------------
__global__ void k_str_scores(const float* __restrict__ hstr,
                             const float* __restrict__ u1, const float* __restrict__ u2,
                             const float* __restrict__ bu,
                             float* __restrict__ evec, float* __restrict__ fvec) {
  const int wave = threadIdx.x >> 5, lane = threadIdx.x & 31;
  const int row = blockIdx.x * 8 + wave;
  const float* h = hstr + (size_t)row * IN_FEAT;
  float x[8];
  float mx = -INFINITY;
#pragma unroll
  for (int t = 0; t < 8; ++t) { x[t] = h[lane + 32 * t]; mx = fmaxf(mx, x[t]); }
  mx = waveMax(mx);
  float s = 0.f, d1 = 0.f, d2 = 0.f;
#pragma unroll
  for (int t = 0; t < 8; ++t) {
    int k = lane + 32 * t;
    float ex = expf(x[t] - mx);
    s += ex; d1 += ex * u1[k]; d2 += ex * u2[k];
  }
  s = waveSum(s); d1 = waveSum(d1); d2 = waveSum(d2);
  if (lane == 0) { evec[row] = d1 / s + bu[0]; fvec[row] = d2 / s + bu[1]; }
}

// ---------------- adjacency bitmap ----------------
__global__ void k_zero(unsigned* __restrict__ p, int n) {
  for (int i = blockIdx.x * blockDim.x + threadIdx.x; i < n; i += gridDim.x * blockDim.x)
    p[i] = 0u;
}
__global__ void k_scatter(const long long* __restrict__ ei, unsigned* __restrict__ adj) {
  int e = blockIdx.x * blockDim.x + threadIdx.x;
  if (e >= N_EDGES) return;
  int i = (int)ei[e];            // edge_index[0, e]
  int j = (int)ei[N_EDGES + e];  // edge_index[1, e]
  atomicOr(&adj[(size_t)i * ADJ_WORDS + (j >> 5)], 1u << (j & 31));
}

// ---------------- column mean of hc (zero-degree fallback) ----------------
__global__ void k_colmean(const float* __restrict__ hc, float* __restrict__ cm) {
  __shared__ float red[256];
  const int col = blockIdx.x;   // 0..127
  float s = 0.f;
  for (int r = threadIdx.x; r < N_NODES; r += 256) s += hc[(size_t)r * OUT_FEAT + col];
  red[threadIdx.x] = s;
  __syncthreads();
  for (int off = 128; off > 0; off >>= 1) {
    if ((int)threadIdx.x < off) red[threadIdx.x] += red[threadIdx.x + off];
    __syncthreads();
  }
  if (threadIdx.x == 0) cm[col] = red[0] / (float)N_NODES;
}

// ---------------- sparse softmax + aggregation: out[i] = sum_j w_ij hc[j] ----------------
// grid = 8192 rows, block = 128 (thread t owns output feature t)
__global__ void k_aggregate(const float* __restrict__ hc,
                            const float* __restrict__ cvec, const float* __restrict__ dvec,
                            const float* __restrict__ evec, const float* __restrict__ fvec,
                            const unsigned* __restrict__ adj,
                            const float* __restrict__ colmean,
                            const float* __restrict__ ws_coff, const float* __restrict__ wc_coff,
                            float* __restrict__ out) {
  const int i = blockIdx.x;
  const int t = threadIdx.x;             // 0..127
  const int wave = t >> 5, lane = t & 31;
  const float A = fabsf(ws_coff[0]);     // multiplies context score (per reference)
  const float B = fabsf(wc_coff[0]);     // multiplies structure score
  const float ci = cvec[i], ei = evec[i];
  const unsigned* row = adj + (size_t)i * ADJ_WORDS;

  __shared__ float red[4];
  __shared__ int   jlist[CAP];
  __shared__ float alist[CAP];
  __shared__ int   cnt;
  __shared__ float mshared, swshared;

  // pass 1: row max of alpha over neighbors
  float mx = -INFINITY;
  for (int w = t; w < ADJ_WORDS; w += 128) {
    unsigned bits = row[w];
    while (bits) {
      int b = __ffs(bits) - 1; bits &= bits - 1;
      int j = (w << 5) + b;
      float a1 = ci + dvec[j]; a1 = a1 > 0.f ? a1 : 0.01f * a1;
      float a2 = ei + fvec[j]; a2 = a2 > 0.f ? a2 : 0.01f * a2;
      mx = fmaxf(mx, A * a1 + B * a2);
    }
  }
  mx = waveMax(mx);
  if (lane == 0) red[wave] = mx;
  __syncthreads();
  if (t == 0) {
    float m = red[0];
    for (int q = 1; q < 4; ++q) m = fmaxf(m, red[q]);
    mshared = m;
  }
  __syncthreads();
  mx = mshared;

  // pass 2: build neighbor list per segment, weighted-accumulate features
  float acc = 0.f, sw = 0.f;
  for (int seg = 0; seg < ADJ_WORDS; seg += SEG_WORDS) {
    if (t == 0) cnt = 0;
    __syncthreads();
    if (t < SEG_WORDS) {
      int w = seg + t;
      unsigned bits = row[w];
      while (bits) {
        int b = __ffs(bits) - 1; bits &= bits - 1;
        int j = (w << 5) + b;
        float a1 = ci + dvec[j]; a1 = a1 > 0.f ? a1 : 0.01f * a1;
        float a2 = ei + fvec[j]; a2 = a2 > 0.f ? a2 : 0.01f * a2;
        int idx = atomicAdd(&cnt, 1);
        jlist[idx] = j;
        alist[idx] = A * a1 + B * a2;
        __builtin_prefetch(hc + (size_t)j * OUT_FEAT, 0, 0);   // global_prefetch_b8
      }
    }
    __syncthreads();
    int n = cnt;
    for (int e = 0; e < n; ++e) {
      float we = expf(alist[e] - mx);
      acc += we * hc[(size_t)jlist[e] * OUT_FEAT + t];
      if ((e & 127) == t) sw += we;   // each weight counted exactly once
    }
    __syncthreads();
  }

  sw = waveSum(sw);
  if (lane == 0) red[wave] = sw;
  __syncthreads();
  if (t == 0) swshared = red[0] + red[1] + red[2] + red[3];
  __syncthreads();
  float total = swshared;
  // zero-degree row: dense softmax over all-NEG row = uniform 1/N -> column mean
  out[(size_t)i * OUT_FEAT + t] = (total > 0.f) ? acc / total : colmean[t];
}

extern "C" void kernel_launch(void* const* d_in, const int* in_sizes, int n_in,
                              void* d_out, int out_size, void* d_ws, size_t ws_size,
                              hipStream_t stream) {
  (void)in_sizes; (void)n_in; (void)out_size; (void)ws_size;
  const float*     h_context = (const float*)d_in[0];
  const float*     h_structure = (const float*)d_in[1];
  const long long* edge_index = (const long long*)d_in[2];   // int64 per reference
  const float*     Wc_w = (const float*)d_in[3];
  const float*     Wc_b = (const float*)d_in[4];
  const float*     Ws_w = (const float*)d_in[5];
  const float*     Ws_b = (const float*)d_in[6];
  const float*     ac_w = (const float*)d_in[7];
  const float*     as_w = (const float*)d_in[8];
  const float*     Ws_coff = (const float*)d_in[9];
  const float*     Wc_coff = (const float*)d_in[10];
  float* out = (float*)d_out;

  char* ws = (char*)d_ws;
  float*    hc   = (float*)ws;    ws += (size_t)N_NODES * OUT_FEAT * sizeof(float);   // 4 MB
  unsigned* adj  = (unsigned*)ws; ws += (size_t)N_NODES * ADJ_WORDS * sizeof(unsigned); // 8 MB
  float*    cvec = (float*)ws;    ws += (size_t)N_NODES * sizeof(float);
  float*    dvec = (float*)ws;    ws += (size_t)N_NODES * sizeof(float);
  float*    evec = (float*)ws;    ws += (size_t)N_NODES * sizeof(float);
  float*    fvec = (float*)ws;    ws += (size_t)N_NODES * sizeof(float);
  float*    u1   = (float*)ws;    ws += (size_t)IN_FEAT * sizeof(float);
  float*    u2   = (float*)ws;    ws += (size_t)IN_FEAT * sizeof(float);
  float*    bu   = (float*)ws;    ws += 2 * sizeof(float);
  float*    cmean= (float*)ws;    ws += (size_t)OUT_FEAT * sizeof(float);

  k_u_vectors<<<1, 256, 0, stream>>>(Ws_w, Ws_b, as_w, u1, u2, bu);
  k_zero<<<2048, 256, 0, stream>>>(adj, N_NODES * ADJ_WORDS);
  k_scatter<<<N_EDGES / 256, 256, 0, stream>>>(edge_index, adj);
  k_gemm_hc<<<dim3(N_NODES / 16, OUT_FEAT / 16), 32, 0, stream>>>(h_context, Wc_w, Wc_b, hc);
  k_ctx_scores<<<N_NODES / 8, 256, 0, stream>>>(hc, ac_w, cvec, dvec);
  k_str_scores<<<N_NODES / 8, 256, 0, stream>>>(h_structure, u1, u2, bu, evec, fvec);
  k_colmean<<<OUT_FEAT, 256, 0, stream>>>(hc, cmean);
  k_aggregate<<<N_NODES, 128, 0, stream>>>(hc, cvec, dvec, evec, fvec, adj, cmean,
                                           Ws_coff, Wc_coff, out);
}
